// Model_47854525612582
// MI455X (gfx1250) — matrix operand order repaired and checked by reading, not compiled, not run on hardware
//
#include <hip/hip_runtime.h>
#include <hip/hip_bf16.h>
#include <cstdint>

#define T_ 512
#define B_ 512
#define D_ 128
#define H_ 256
#define C_ 1000
#define CP_ 1008          // C padded to multiple of 16 for WMMA N-tiles
#define LDH 264           // padded LDS row stride (bf16 elems): 528B -> conflict-free
#define HBS (16 * LDH)    // one h buffer (16 rows)

typedef __attribute__((ext_vector_type(16))) __bf16 v16bf;
typedef __attribute__((ext_vector_type(8)))  float  v8f;

// D = A(16x32 bf16) * B(32x16 bf16) + C(16x16 f32)
__device__ __forceinline__ v8f wmma_bf16(v16bf a, v16bf b, v8f c) {
  return __builtin_amdgcn_wmma_f32_16x16x32_bf16(
      /*neg_a=*/false, a, /*neg_b=*/false, b, /*c_mod=*/(short)0, c,
      /*reuse_a=*/false, /*reuse_b=*/false);
}

// A fragment (16x32, MxK) from row-major [16 x ldk] tile.
// ISA: lane<16 holds row M=lane, K={0..7,16..23}; lane>=16: K={8..15,24..31}
__device__ __forceinline__ v16bf load_frag_a(const __bf16* base, int ldk, int lane) {
  int r = lane & 15, g = lane >> 4;
  const __bf16* p = base + (size_t)r * ldk + g * 8;
  v16bf a;
  *((uint4*)&a)     = *(const uint4*)p;
  *(((uint4*)&a)+1) = *(const uint4*)(p + 16);
  return a;
}

// B fragment (32x16, KxN): column n of B = row n of row-major W[N x ldk].
// ISA: lane<16 holds col N=lane&15, K={0..7,16..23}; lane>=16: K={8..15,24..31}
__device__ __forceinline__ v16bf load_frag_b(const __bf16* base, int ldk, int lane) {
  int n = lane & 15, g = lane >> 4;
  const __bf16* p = base + (size_t)n * ldk + g * 8;
  v16bf b;
  *((uint4*)&b)     = *(const uint4*)p;
  *(((uint4*)&b)+1) = *(const uint4*)(p + 16);
  return b;
}

// cheap device sigmoid: 1 v_exp + 1 v_add + 1 v_rcp (+1 mul for scaling)
__device__ __forceinline__ float sigm(float x) {
  return __builtin_amdgcn_rcpf(1.f + __expf(-x));
}

// ---- prologue: f32 -> bf16 weights (W_out zero-padded to 1008 rows)
__global__ void cvt_w_kernel(const float* __restrict__ wih, const float* __restrict__ whh,
                             const float* __restrict__ wout,
                             __bf16* __restrict__ wihb, __bf16* __restrict__ whhb,
                             __bf16* __restrict__ woutb) {
  int i = blockIdx.x * blockDim.x + threadIdx.x;
  const int N1 = 3 * H_ * D_;
  const int N2 = 3 * H_ * H_;
  const int N3 = CP_ * H_;
  if (i < N1) wihb[i] = (__bf16)wih[i];
  if (i < N2) whhb[i] = (__bf16)whh[i];
  if (i < N3) woutb[i] = (__bf16)((i < C_ * H_) ? wout[i] : 0.f);
}

// ---- prologue: x (B,T,D) f32 -> xt (T,B,D) bf16 ; both sides coalesced (d innermost)
__global__ void cvt_x_kernel(const float* __restrict__ x, __bf16* __restrict__ xt) {
  size_t i = (size_t)blockIdx.x * blockDim.x + threadIdx.x;
  if (i >= (size_t)T_ * B_ * D_) return;
  int d = (int)(i % D_);
  size_t tb = i / D_;
  int b = (int)(tb % B_);
  int t = (int)(tb / B_);
  xt[i] = (__bf16)x[((size_t)b * T_ + t) * D_ + d];
}

// ---- sequential GRU scan: 1 block = 16 waves = 16 batch rows, full T loop on-chip.
// Wave w owns hidden cols [16w,16w+16): tiles {w,w+16,w+32} are r/z/n for those cols.
// Weight placement: W_hh[z],W_hh[n] -> registers (128 VGPR), W_hh[r] -> LDS, W_ih -> L2.
__global__ __launch_bounds__(512) void gru_seq_kernel(
    const __bf16* __restrict__ xt,     // (T,B,D) bf16
    const __bf16* __restrict__ wihb,   // (3H,D) bf16
    const __bf16* __restrict__ whhb,   // (3H,H) bf16
    const float*  __restrict__ b_ih,
    const float*  __restrict__ b_hh,
    const float*  __restrict__ dmask,  // (T,B,H) f32
    __bf16* __restrict__ hfin)         // (B,H) bf16
{
  extern __shared__ __align__(16) char smem_raw[];
  __bf16* wlds = (__bf16*)smem_raw;        // [256][LDH]  W_hh r-gate, padded rows
  __bf16* hb   = wlds + 256 * LDH;         // [2][16][LDH] double-buffered h

  const int tid  = threadIdx.x;
  const int lane = tid & 31, wave = tid >> 5;   // 16 waves
  const int brow0 = blockIdx.x * 16;
  const int nc    = wave * 16;
  const int col   = lane & 15, g = lane >> 4;

  // stage W_hh r-gate (rows 0..255) into padded LDS with CDNA5 async global->LDS DMA.
  // Per-lane: LDS[vdst] <- MEM[vaddr], 16B each; tracked by ASYNCcnt (no VGPR round-trip).
  for (int i = tid; i < 256 * 32; i += 512) {
    int row = i >> 5, c = i & 31;
    unsigned ldsb = (unsigned)(uintptr_t)(void*)(wlds + row * LDH + c * 8);  // LDS byte addr
    const __bf16* gsrc = whhb + (size_t)row * H_ + c * 8;
    asm volatile("global_load_async_to_lds_b128 %0, %1, off"
                 :: "v"(ldsb), "v"(gsrc) : "memory");
  }
  for (int i = tid; i < HBS; i += 512) hb[i] = (__bf16)0.f;  // zero h buffer 0

  // W_hh z-gate and n-gate fragments: register-resident for the whole scan
  v16bf wz[8], wn[8];
#pragma unroll
  for (int k = 0; k < 8; k++) {
    wz[k] = load_frag_b(whhb + (size_t)(1 * H_ + nc) * H_ + k * 32, H_, lane);
    wn[k] = load_frag_b(whhb + (size_t)(2 * H_ + nc) * H_ + k * 32, H_, lane);
  }
  const float br  = b_ih[0 * H_ + nc + col] + b_hh[0 * H_ + nc + col];
  const float bz  = b_ih[1 * H_ + nc + col] + b_hh[1 * H_ + nc + col];
  const float bin = b_ih[2 * H_ + nc + col];
  const float bhn = b_hh[2 * H_ + nc + col];

  asm volatile("s_wait_asynccnt 0x0" ::: "memory");  // async LDS fill complete (this wave)
  __syncthreads();                                   // ... and visible workgroup-wide

  unsigned blg = 0, bll = 0;  // opaque zero offsets: defeat LICM on streamed loads
  for (int t = 0; t < T_; ++t) {
    asm volatile("" : "+v"(blg), "+v"(bll));
    const __bf16* hcur = hb + (t & 1) * HBS;
    __bf16*       hnxt = hb + ((t & 1) ^ 1) * HBS;
    const __bf16* xrow = xt + ((size_t)t * B_ + brow0) * D_;

    v8f ar = {}, az = {}, ain = {}, ahn = {};
    // gi part: x A-frags + W_ih B-frags streamed (L2-resident; blg keeps them in-loop)
#pragma unroll
    for (int k = 0; k < 4; k++) {
      v16bf xk = load_frag_a(xrow + k * 32, D_, lane);
      ar  = wmma_bf16(xk, load_frag_b(wihb + blg + (size_t)(0 * H_ + nc) * D_ + k * 32, D_, lane), ar);
      az  = wmma_bf16(xk, load_frag_b(wihb + blg + (size_t)(1 * H_ + nc) * D_ + k * 32, D_, lane), az);
      ain = wmma_bf16(xk, load_frag_b(wihb + blg + (size_t)(2 * H_ + nc) * D_ + k * 32, D_, lane), ain);
    }
    // gh part: h A-frags from LDS; r-weights from LDS, z/n-weights from registers
#pragma unroll
    for (int k = 0; k < 8; k++) {
      v16bf hk = load_frag_a(hcur + k * 32, LDH, lane);
      ar  = wmma_bf16(hk, load_frag_b(wlds + bll + (size_t)nc * LDH + k * 32, LDH, lane), ar);
      az  = wmma_bf16(hk, wz[k], az);
      ahn = wmma_bf16(hk, wn[k], ahn);
    }

    const float* mrow = dmask + ((size_t)t * B_ + brow0) * H_;
#pragma unroll
    for (int v = 0; v < 8; v++) {
      int r = v + 8 * g;
      float rg = sigm(ar[v] + br);
      float zg = sigm(az[v] + bz);
      // tanh(a) = 2*sigmoid(2a) - 1 : exp-based, no ocml polynomial
      float na = ain[v] + bin + rg * (ahn[v] + bhn);
      float ng = 2.f * sigm(2.f * na) - 1.f;
      float hold = (float)hcur[r * LDH + nc + col];  // h_old from LDS (bf16 image of h)
      float hn = (1.f - zg) * ng + zg * hold;
      hn *= mrow[(size_t)r * H_ + nc + col];
      hnxt[r * LDH + nc + col] = (__bf16)hn;  // write to the *other* buffer
    }
    __syncthreads();  // single barrier: h_new visible before next step's reads
  }

  // T_ is even: final h lives in buffer 0 (= hb); plain bf16 copy out
#pragma unroll
  for (int v = 0; v < 8; v++) {
    int r = v + 8 * g;
    hfin[(size_t)(brow0 + r) * H_ + nc + col] = hb[r * LDH + nc + col];
  }
}

// ---- epilogue GEMM: out(512x1000) = hfin(512x256) @ W_out^T + b_out
__global__ __launch_bounds__(256) void out_gemm_kernel(
    const __bf16* __restrict__ hfin, const __bf16* __restrict__ woutb,
    const float* __restrict__ b_out, float* __restrict__ out) {
  int lane = threadIdx.x & 31;
  int wave = threadIdx.x >> 5;
  int tile = blockIdx.x * 8 + wave;      // 32 * 63 = 2016 tiles, 252 blocks x 8 waves
  int mt = tile / (CP_ / 16);
  int nt = tile % (CP_ / 16);
  const __bf16* abase = hfin  + (size_t)mt * 16 * H_;
  const __bf16* bbase = woutb + (size_t)nt * 16 * H_;
  v8f acc = {};
#pragma unroll
  for (int k = 0; k < 8; k++) {
    v16bf a = load_frag_a(abase + k * 32, H_, lane);
    v16bf b = load_frag_b(bbase + k * 32, H_, lane);
    acc = wmma_bf16(a, b, acc);
  }
  int col = lane & 15, g = lane >> 4;
  int n = nt * 16 + col;
  if (n < C_) {
    float bb = b_out[n];
#pragma unroll
    for (int v = 0; v < 8; v++) {
      int row = mt * 16 + v + 8 * g;
      out[(size_t)row * C_ + n] = acc[v] + bb;
    }
  }
}

extern "C" void kernel_launch(void* const* d_in, const int* in_sizes, int n_in,
                              void* d_out, int out_size, void* d_ws, size_t ws_size,
                              hipStream_t stream) {
  (void)in_sizes; (void)n_in; (void)out_size; (void)ws_size;
  const float* x     = (const float*)d_in[0];
  const float* W_ih  = (const float*)d_in[1];
  const float* W_hh  = (const float*)d_in[2];
  const float* b_ih  = (const float*)d_in[3];
  const float* b_hh  = (const float*)d_in[4];
  const float* W_out = (const float*)d_in[5];
  const float* b_out = (const float*)d_in[6];
  const float* dmask = (const float*)d_in[7];
  float* out = (float*)d_out;

  char* ws = (char*)d_ws;
  size_t off = 0;
  auto alloc = [&](size_t bytes) -> char* {
    char* p = ws + off;
    off += (bytes + 255) & ~(size_t)255;
    return p;
  };
  __bf16* xt    = (__bf16*)alloc((size_t)T_ * B_ * D_ * 2);  // 67 MB
  __bf16* wihb  = (__bf16*)alloc((size_t)3 * H_ * D_ * 2);
  __bf16* whhb  = (__bf16*)alloc((size_t)3 * H_ * H_ * 2);
  __bf16* woutb = (__bf16*)alloc((size_t)CP_ * H_ * 2);
  __bf16* hfin  = (__bf16*)alloc((size_t)B_ * H_ * 2);

  cvt_w_kernel<<<(CP_ * H_ + 255) / 256, 256, 0, stream>>>(W_ih, W_hh, W_out, wihb, whhb, woutb);
  size_t nx = (size_t)T_ * B_ * D_;
  cvt_x_kernel<<<(int)((nx + 255) / 256), 256, 0, stream>>>(x, xt);

  // dynamic LDS: W_hh r-gate (padded) + double-buffered h = 152,064 bytes (<320KB/WGP)
  const size_t smem = (size_t)(256 * LDH + 2 * HBS) * sizeof(__bf16);
  gru_seq_kernel<<<B_ / 16, 512, smem, stream>>>(xt, wihb, whhb, b_ih, b_hh, dmask, hfin);

  out_gemm_kernel<<<(32 * (CP_ / 16)) / 8, 256, 0, stream>>>(hfin, woutb, b_out, out);
}
